// Attention_14474039787737
// MI455X (gfx1250) — compile-verified, matrix operand order checked by reading, not run on hardware
//
#include <hip/hip_runtime.h>
#include <hip/hip_bf16.h>

// MHA: B=2, S=2048, D=2048, H=16, HD=128.
// Compute-bound (~206 GFLOP vs ~280MB min traffic) -> all matmuls on
// v_wmma_f32_16x16x32_bf16 (f32 accum). Flash-attention avoids the 536MB
// score tensor. Workspace: Q,K,V bf16 [B,H,S,HD] (3x16MB) + O fp32 (32MB).

#define BATCH  2
#define SEQ    2048
#define DMODEL 2048
#define NHEAD  16
#define HDIM   128
#define MROWS  (BATCH * SEQ)   // 4096

typedef __attribute__((ext_vector_type(16))) __bf16 v16bf;
typedef __attribute__((ext_vector_type(8)))  float  v8f;
typedef __attribute__((ext_vector_type(4)))  int    v4i;

typedef __attribute__((address_space(1))) v4i as1_v4i;
typedef __attribute__((address_space(3))) v4i as3_v4i;

union FragU { uint4 u[2]; v16bf v; };

__device__ __forceinline__ unsigned short f2bf(float x) {
    unsigned int u = __float_as_uint(x);
    return (unsigned short)((u + 0x7FFFu + ((u >> 16) & 1u)) >> 16);  // RNE
}

// Pack two floats -> packed bf16 pair (v_cvt_pk_bf16_f32 when available).
__device__ __forceinline__ unsigned int pack2bf(float x, float y) {
#if __has_builtin(__builtin_amdgcn_cvt_pk_bf16_f32)
    auto r = __builtin_amdgcn_cvt_pk_bf16_f32(x, y);
    unsigned int u;
    __builtin_memcpy(&u, &r, 4);
    return u;
#else
    return (unsigned)f2bf(x) | ((unsigned)f2bf(y) << 16);
#endif
}

// Async global->LDS 16B copy (ASYNCcnt path) with plain-copy fallback.
__device__ __forceinline__ void async_cp16(void* lds_dst, const void* gsrc) {
#if __has_builtin(__builtin_amdgcn_global_load_async_to_lds_b128)
    __builtin_amdgcn_global_load_async_to_lds_b128(
        (as1_v4i*)gsrc, (as3_v4i*)lds_dst, 0, 0);
#else
    *(uint4*)lds_dst = *(const uint4*)gsrc;
#endif
}
__device__ __forceinline__ void wait_async0() {
#if __has_builtin(__builtin_amdgcn_s_wait_asynccnt)
    __builtin_amdgcn_s_wait_asynccnt(0);
#else
    asm volatile("s_wait_asynccnt 0x0" ::: "memory");
#endif
}

__device__ __forceinline__ v8f zero8() {
    v8f z;
#pragma unroll
    for (int e = 0; e < 8; ++e) z[e] = 0.0f;
    return z;
}

// Load one 16x32 bf16 A/B fragment (ISA 16-bit A layout):
// lanes 0-15 take K[0..7],K[16..23]; lanes 16-31 take K[8..15],K[24..31].
__device__ __forceinline__ v16bf ld_frag(const unsigned short* rowp, int half) {
    FragU f;
    f.u[0] = *(const uint4*)(rowp + half * 8);
    f.u[1] = *(const uint4*)(rowp + 16 + half * 8);
    return f.v;
}

__device__ __forceinline__ float rmax16(float v) {
#pragma unroll
    for (int m = 1; m < 16; m <<= 1) v = fmaxf(v, __shfl_xor(v, m, 32));
    return v;
}
__device__ __forceinline__ float rsum16(float v) {
#pragma unroll
    for (int m = 1; m < 16; m <<= 1) v += __shfl_xor(v, m, 32);
    return v;
}

// ---------------------------------------------------------------------------
// Y = A[M,K] @ W[N,K]^T + bias[N].  fp32 sources, bf16 WMMA, f32 accum.
// HEADSPLIT: bf16 out into [B,H,S,HD]; else fp32 out into [M,N].
// Block: 256 thr (8 waves), tile 128x128, BK=64 (16 WMMA/wave per barrier).
// ---------------------------------------------------------------------------
template <bool HEADSPLIT>
__global__ __launch_bounds__(256) void proj_gemm(
    const float* __restrict__ A, const float* __restrict__ W,
    const float* __restrict__ bias,
    unsigned short* __restrict__ outBf, float* __restrict__ outF)
{
    const int K = DMODEL, N = DMODEL;
    __shared__ alignas(16) unsigned short As[128][64];
    __shared__ alignas(16) unsigned short Bs[128][64];

    const int tid  = threadIdx.x;
    const int lane = tid & 31, wave = tid >> 5;
    const int half = lane >> 4, l16 = lane & 15;
    const int wm = wave & 3, wn = wave >> 2;  // 4x2 wave grid: 32 rows x 64 cols
    const int m0 = blockIdx.y * 128, n0 = blockIdx.x * 128;

    v8f acc[2][4];
#pragma unroll
    for (int i = 0; i < 2; ++i)
#pragma unroll
        for (int j = 0; j < 4; ++j) acc[i][j] = zero8();

    for (int kt = 0; kt < K; kt += 64) {
        // Cooperative load + fp32->bf16 pack into LDS (row-major [row][k]).
#pragma unroll
        for (int i = 0; i < 8; ++i) {
            int idx = tid + i * 256;                  // 0..2047 float4's per tile
            int r = idx >> 4, c = (idx & 15) << 2;
            float4 a4 = *(const float4*)(A + (size_t)(m0 + r) * K + kt + c);
            *(uint2*)&As[r][c] = make_uint2(pack2bf(a4.x, a4.y), pack2bf(a4.z, a4.w));
            float4 b4 = *(const float4*)(W + (size_t)(n0 + r) * K + kt + c);
            *(uint2*)&Bs[r][c] = make_uint2(pack2bf(b4.x, b4.y), pack2bf(b4.z, b4.w));
            if (i == 0 && kt + 64 < K) {
                __builtin_prefetch(A + (size_t)(m0 + r) * K + kt + 64 + c, 0, 1);
                __builtin_prefetch(W + (size_t)(n0 + r) * K + kt + 64 + c, 0, 1);
            }
        }
        __syncthreads();

#pragma unroll
        for (int c = 0; c < 2; ++c) {
            v16bf af[2], bfr[4];
#pragma unroll
            for (int i = 0; i < 2; ++i)
                af[i] = ld_frag(&As[wm * 32 + i * 16 + l16][c * 32], half);
#pragma unroll
            for (int j = 0; j < 4; ++j)
                bfr[j] = ld_frag(&Bs[wn * 64 + j * 16 + l16][c * 32], half);
#pragma unroll
            for (int i = 0; i < 2; ++i)
#pragma unroll
                for (int j = 0; j < 4; ++j)
                    acc[i][j] = __builtin_amdgcn_wmma_f32_16x16x32_bf16(
                        false, af[i], false, bfr[j], (short)0, acc[i][j], false, false);
        }
        __syncthreads();
    }

    // Epilogue: bias + store. C layout: vgpr r -> M = r + 8*half, lane l16 -> N.
#pragma unroll
    for (int i = 0; i < 2; ++i) {
#pragma unroll
        for (int j = 0; j < 4; ++j) {
            int gcol = n0 + wn * 64 + j * 16 + l16;
            float bv = bias[gcol];
#pragma unroll
            for (int r = 0; r < 8; ++r) {
                int grow = m0 + wm * 32 + i * 16 + r + half * 8;
                float v = acc[i][j][r] + bv;
                if (HEADSPLIT) {
                    int b  = grow >> 11;          // / SEQ
                    int s  = grow & (SEQ - 1);
                    int h  = gcol >> 7;           // / HDIM
                    int hd = gcol & (HDIM - 1);
                    size_t o = ((size_t)(b * NHEAD + h) * SEQ + s) * HDIM + hd;
                    outBf[o] = f2bf(v);
                } else {
                    outF[(size_t)grow * N + gcol] = v;
                }
            }
        }
    }
}

// ---------------------------------------------------------------------------
// Flash attention over one (q-block of 128, head, batch). KV blocks of 64.
// Q fragments resident in registers; K tile fetched with async global->LDS;
// V stored transposed in LDS (packed b32 stores); P relaid via wave-private LDS.
// ---------------------------------------------------------------------------
__global__ __launch_bounds__(256) void flash_attn(
    const unsigned short* __restrict__ Qb, const unsigned short* __restrict__ Kb,
    const unsigned short* __restrict__ Vb, float* __restrict__ O)
{
    __shared__ alignas(16) unsigned short Ks[64][128];    // [kv row][hd]
    __shared__ alignas(16) unsigned short Vs[128][64];    // transposed: [hd][kv row]
    __shared__ alignas(16) unsigned short Ps[8][16][64];  // per-wave P tile

    const int tid  = threadIdx.x;
    const int lane = tid & 31, wave = tid >> 5;
    const int half = lane >> 4, l16 = lane & 15;
    const int q0 = blockIdx.x * 128;
    const int h = blockIdx.y, b = blockIdx.z;
    const size_t base = (size_t)(b * NHEAD + h) * SEQ * HDIM;
    const unsigned short* Q  = Qb + base;
    const unsigned short* Kp = Kb + base;
    const unsigned short* Vp = Vb + base;

    // Resident Q fragments: 16 rows x 128 HD per wave, 4 K-chunks of 32.
    v16bf qf[4];
    {
        const unsigned short* qrow = Q + (size_t)(q0 + wave * 16 + l16) * HDIM;
#pragma unroll
        for (int c = 0; c < 4; ++c) qf[c] = ld_frag(qrow + c * 32, half);
    }

    v8f oacc[8];
#pragma unroll
    for (int n = 0; n < 8; ++n) oacc[n] = zero8();
    float mrow[8], lrow[8];
#pragma unroll
    for (int r = 0; r < 8; ++r) { mrow[r] = -INFINITY; lrow[r] = 0.0f; }

    const float kscale = 0.08838834764831845f * 1.4426950408889634f;  // 1/sqrt(128)*log2(e)

    for (int kv0 = 0; kv0 < SEQ; kv0 += 64) {
        // K tile: async bf16 copy straight into LDS (ASYNCcnt path).
#pragma unroll
        for (int i = 0; i < 4; ++i) {
            int idx = tid + i * 256;                  // 1024 uint4's
            int r = idx >> 4, c = (idx & 15) << 3;
            async_cp16(&Ks[r][c], Kp + (size_t)(kv0 + r) * HDIM + c);
        }
        // V tile, transposed into Vs[hd][kv]: two kv rows packed per b32 store.
#pragma unroll
        for (int i = 0; i < 2; ++i) {
            int idx = tid + i * 256;                  // 0..511
            int rp = (idx >> 4) << 1;                 // even kv row
            int c  = (idx & 15) << 3;
            uint4 v0 = *(const uint4*)(Vp + (size_t)(kv0 + rp) * HDIM + c);
            uint4 v1 = *(const uint4*)(Vp + (size_t)(kv0 + rp + 1) * HDIM + c);
            const unsigned short* e0 = (const unsigned short*)&v0;
            const unsigned short* e1 = (const unsigned short*)&v1;
#pragma unroll
            for (int j = 0; j < 8; ++j)
                *(unsigned int*)&Vs[c + j][rp] =
                    (unsigned)e0[j] | ((unsigned)e1[j] << 16);
        }
        wait_async0();
        __syncthreads();

        // Scores: S(16x64) = Q(16x128) @ K_block^T.
        v8f sacc[4];
#pragma unroll
        for (int j = 0; j < 4; ++j) sacc[j] = zero8();
#pragma unroll
        for (int c = 0; c < 4; ++c)
#pragma unroll
            for (int j = 0; j < 4; ++j) {
                v16bf kf = ld_frag(&Ks[j * 16 + l16][c * 32], half);
                sacc[j] = __builtin_amdgcn_wmma_f32_16x16x32_bf16(
                    false, qf[c], false, kf, (short)0, sacc[j], false, false);
            }

        // Online softmax (row = vgpr r + 8*half; 16-lane cross reductions).
        float alpha[8], nm[8];
#pragma unroll
        for (int r = 0; r < 8; ++r) {
            float mx = sacc[0][r];
#pragma unroll
            for (int j = 1; j < 4; ++j) mx = fmaxf(mx, sacc[j][r]);
            mx = rmax16(mx);
            nm[r] = fmaxf(mrow[r], mx);
            alpha[r] = exp2f((mrow[r] - nm[r]) * kscale);
            mrow[r] = nm[r];
        }
        float rs[8];
#pragma unroll
        for (int r = 0; r < 8; ++r) rs[r] = 0.0f;
#pragma unroll
        for (int j = 0; j < 4; ++j)
#pragma unroll
            for (int r = 0; r < 8; ++r) {
                float p = exp2f((sacc[j][r] - nm[r]) * kscale);
                sacc[j][r] = p;
                rs[r] += p;
            }
#pragma unroll
        for (int r = 0; r < 8; ++r) {
            lrow[r] = lrow[r] * alpha[r] + rsum16(rs[r]);
#pragma unroll
            for (int n = 0; n < 8; ++n) oacc[n][r] *= alpha[r];
        }

        // Re-layout P (C-layout regs) -> A-layout via wave-private LDS tile.
#pragma unroll
        for (int j = 0; j < 4; ++j)
#pragma unroll
            for (int r = 0; r < 8; ++r)
                Ps[wave][r + half * 8][j * 16 + l16] = f2bf(sacc[j][r]);
        __syncthreads();

        // O(16x128) += P(16x64) @ V(64x128).
#pragma unroll
        for (int c = 0; c < 2; ++c) {
            v16bf pf = ld_frag(&Ps[wave][l16][c * 32], half);
#pragma unroll
            for (int n = 0; n < 8; ++n) {
                v16bf vf = ld_frag(&Vs[n * 16 + l16][c * 32], half);
                oacc[n] = __builtin_amdgcn_wmma_f32_16x16x32_bf16(
                    false, pf, false, vf, (short)0, oacc[n], false, false);
            }
        }
        __syncthreads();
    }

    // Normalize and merge heads: O fp32 in [B,S,D].
#pragma unroll
    for (int n = 0; n < 8; ++n) {
        int col = h * HDIM + n * 16 + l16;
#pragma unroll
        for (int r = 0; r < 8; ++r) {
            int s = q0 + wave * 16 + r + half * 8;
            O[(size_t)(b * SEQ + s) * DMODEL + col] = oacc[n][r] / lrow[r];
        }
    }
}

// ---------------------------------------------------------------------------
extern "C" void kernel_launch(void* const* d_in, const int* in_sizes, int n_in,
                              void* d_out, int out_size, void* d_ws, size_t ws_size,
                              hipStream_t stream) {
    const float* query = (const float*)d_in[0];
    const float* key_  = (const float*)d_in[1];
    const float* value = (const float*)d_in[2];
    const float* Wq = (const float*)d_in[3];
    const float* bq = (const float*)d_in[4];
    const float* Wk = (const float*)d_in[5];
    const float* bk = (const float*)d_in[6];
    const float* Wv = (const float*)d_in[7];
    const float* bv = (const float*)d_in[8];
    const float* Wo = (const float*)d_in[9];
    const float* bo = (const float*)d_in[10];

    // Workspace: Q,K,V bf16 head-split (16MB each) + O fp32 (32MB) = 80MB.
    const size_t QKV = (size_t)BATCH * NHEAD * SEQ * HDIM;
    unsigned short* q = (unsigned short*)d_ws;
    unsigned short* k = q + QKV;
    unsigned short* v = k + QKV;
    float* o = (float*)(v + QKV);

    dim3 blk(256);
    dim3 gproj(DMODEL / 128, MROWS / 128);   // 16 x 32
    proj_gemm<true><<<gproj, blk, 0, stream>>>(query, Wq, bq, q, nullptr);
    proj_gemm<true><<<gproj, blk, 0, stream>>>(key_, Wk, bk, k, nullptr);
    proj_gemm<true><<<gproj, blk, 0, stream>>>(value, Wv, bv, v, nullptr);

    dim3 gfa(SEQ / 128, NHEAD, BATCH);       // 16 x 16 x 2
    flash_attn<<<gfa, blk, 0, stream>>>(q, k, v, o);

    proj_gemm<false><<<gproj, blk, 0, stream>>>(o, Wo, bo, nullptr, (float*)d_out);
}
